// TensorizedCircuit_24352464569751
// MI455X (gfx1250) — matrix operand order repaired and checked by reading, not compiled, not run on hardware
//
#include <hip/hip_runtime.h>
#include <math.h>

typedef __attribute__((ext_vector_type(16))) _Float16 v16h;
typedef __attribute__((ext_vector_type(8)))  float    v8f;

// Problem constants
#define BDIM 4096
#define DDIM 256
#define KDIM 64
#define NSUB 16            // phase-1 subtrees (16 leaves each -> levels 0..3)
#define LEAVES_PER_SUB 16
#define NTILES_B (BDIM / 16)   // 256 batch tiles of M=16

// Workspace layout (bytes)
//  WFRAG : 255 pairs * 2 children * 4 N-tiles * 2 K-chunks * 1024B = 4,177,920
//  GA    : exp(-log_sigma)            D*K f32 = 64KB
//  GB    : -log_sigma - 0.5*log(2pi)  D*K f32 = 64KB
//  LW    : log_softmax(w_root)        64  f32
//  INTER : (B, NSUB, K) f32 = 16MB
#define WFRAG_OFF 0
#define GA_OFF    4194304
#define GB_OFF    (GA_OFF + 65536)
#define LW_OFF    (GA_OFF + 131072)
#define INTER_OFF (GA_OFF + 131072 + 4096)

#define HALF_LOG2PI 0.9189385332046727f

struct W8 { const float* p[8]; };

// level offset into the flat list of 255 combine units: off(l) = 256 - (256>>l)
__device__ __forceinline__ int lvl_off(int l) { return 256 - (256 >> l); }

// ---------------------------------------------------------------------------
// Prep kernels (tiny; run every launch, deterministic)
// ---------------------------------------------------------------------------

// One block per (unit u, child c); thread o in [0,64): softmax row -> f16,
// scattered into WMMA B-fragment layout:
//   fragment f = ((u*2+c)*4 + t)*2 + q   (t = o/16 N-tile, q = k/32 K-chunk)
//   lane ln = (o&15) + 16*((k&31)>>4), element idx = k&15  (2 halves per dword)
__global__ void prep_weights(W8 w, _Float16* __restrict__ wf) {
  const int u = blockIdx.x >> 1;        // 0..254
  const int c = blockIdx.x & 1;
  const int o = threadIdx.x;            // 0..63
  int l = 0;
  while (l < 7 && u >= (256 - (256 >> (l + 1)))) ++l;
  const int r = u - (256 - (256 >> l));
  const float* row = w.p[l] + ((size_t)(r * 2 + c) * 64 + o) * 64;

  float mx = -INFINITY;
  for (int k = 0; k < 64; ++k) mx = fmaxf(mx, row[k]);
  float s = 0.f;
  for (int k = 0; k < 64; ++k) s += __expf(row[k] - mx);
  const float lse = mx + __logf(s);

  const int t = o >> 4, n = o & 15;
  for (int k = 0; k < 64; ++k) {
    const float v = __expf(row[k] - lse);       // softmax in [0,1] -> safe f16
    const int q  = k >> 5;
    const int kr = k & 31;
    const int ln = n + ((kr >> 4) << 4);
    const int kl = kr & 15;
    const size_t f = (size_t)(((u * 2 + c) * 4 + t) * 2 + q);
    wf[f * 512 + (size_t)ln * 16 + kl] = (_Float16)v;
  }
}

__global__ void prep_gauss(const float* __restrict__ ls,
                           float* __restrict__ ga, float* __restrict__ gb) {
  const int i = blockIdx.x * 256 + threadIdx.x;   // D*K = 16384
  const float v = ls[i];
  ga[i] = __expf(-v);
  gb[i] = -v - HALF_LOG2PI;
}

__global__ void prep_root(const float* __restrict__ wr, float* __restrict__ lw) {
  const int t = threadIdx.x;  // 64 threads, redundant reduction (trivial)
  float mx = -INFINITY;
  for (int k = 0; k < 64; ++k) mx = fmaxf(mx, wr[k]);
  float s = 0.f;
  for (int k = 0; k < 64; ++k) s += __expf(wr[k] - mx);
  lw[t] = wr[t] - (mx + __logf(s));
}

// ---------------------------------------------------------------------------
// Core combine: (h0, h1) 16x64 f32 log-values in LDS  ->  parent 16x64 in dst.
//   per child: m = rowmax; E = exp(h-m) (f16); S = E @ W^T (f32 WMMA accum)
//   parent    = log(S0*S1) + m0 + m1
// Must be called by all 32 lanes, uniform control flow (WMMA needs EXEC=~0).
// ---------------------------------------------------------------------------
__device__ __forceinline__ void combine32(
    float* __restrict__ h0, float* __restrict__ h1, float* __restrict__ dst,
    float* __restrict__ mbuf /*[32]*/, float* __restrict__ msum /*[16]*/,
    const uint4* __restrict__ wf4, int u, int lane)
{
  const int row = lane & 15;
  const int kh  = lane >> 4;         // doubles as child id for the max pass

  __syncthreads();                   // prior writers of h0/h1 visible

  // row-max: lanes 0-15 -> child0 rows, lanes 16-31 -> child1 rows
  {
    const float* hc = kh ? h1 : h0;
    const float4* hr = (const float4*)(hc + row * 64);
    float mx = -INFINITY;
#pragma unroll
    for (int i = 0; i < 16; ++i) {
      float4 v = hr[i];
      mx = fmaxf(mx, fmaxf(fmaxf(v.x, v.y), fmaxf(v.z, v.w)));
    }
    mbuf[kh * 16 + row] = mx;
  }
  __syncthreads();
  if (lane < 16) msum[row] = mbuf[row] + mbuf[16 + row];

  // Build A fragments (16-bit A 16x32, contiguous k order as in prep_weights):
  //   lane = M + 16*khalf; elem i -> K = 32q + 16*kh + i
  v16h afrag[2][2];
#pragma unroll
  for (int cc = 0; cc < 2; ++cc) {
    const float* hs = cc ? h1 : h0;
    const float mv = mbuf[cc * 16 + row];
#pragma unroll
    for (int q = 0; q < 2; ++q) {
      const float4* b4 = (const float4*)(hs + row * 64 + q * 32 + kh * 16);
      float4 v0 = b4[0], v1 = b4[1], v2 = b4[2], v3 = b4[3];
      afrag[cc][q][0]  = (_Float16)__expf(v0.x - mv);
      afrag[cc][q][1]  = (_Float16)__expf(v0.y - mv);
      afrag[cc][q][2]  = (_Float16)__expf(v0.z - mv);
      afrag[cc][q][3]  = (_Float16)__expf(v0.w - mv);
      afrag[cc][q][4]  = (_Float16)__expf(v1.x - mv);
      afrag[cc][q][5]  = (_Float16)__expf(v1.y - mv);
      afrag[cc][q][6]  = (_Float16)__expf(v1.z - mv);
      afrag[cc][q][7]  = (_Float16)__expf(v1.w - mv);
      afrag[cc][q][8]  = (_Float16)__expf(v2.x - mv);
      afrag[cc][q][9]  = (_Float16)__expf(v2.y - mv);
      afrag[cc][q][10] = (_Float16)__expf(v2.z - mv);
      afrag[cc][q][11] = (_Float16)__expf(v2.w - mv);
      afrag[cc][q][12] = (_Float16)__expf(v3.x - mv);
      afrag[cc][q][13] = (_Float16)__expf(v3.y - mv);
      afrag[cc][q][14] = (_Float16)__expf(v3.z - mv);
      afrag[cc][q][15] = (_Float16)__expf(v3.w - mv);
    }
  }
  __syncthreads();                   // all reads of h0/h1 done (dst aliases h0)

  float ms[8];
#pragma unroll
  for (int j = 0; j < 8; ++j) ms[j] = msum[8 * kh + j];   // M = j + 8*khalf

#pragma unroll
  for (int t = 0; t < 4; ++t) {      // N-tiles of 16 output channels
    v8f acc0 = {};
    v8f acc1 = {};
#pragma unroll
    for (int q = 0; q < 2; ++q) {
      union { uint4 u[2]; v16h h; } b0, b1;
      const uint4* w0 = wf4 + (size_t)(((u * 2 + 0) * 4 + t) * 2 + q) * 64 + lane * 2;
      const uint4* w1 = wf4 + (size_t)(((u * 2 + 1) * 4 + t) * 2 + q) * 64 + lane * 2;
      b0.u[0] = w0[0]; b0.u[1] = w0[1];
      b1.u[0] = w1[0]; b1.u[1] = w1[1];
      acc0 = __builtin_amdgcn_wmma_f32_16x16x32_f16(false, afrag[0][q], false, b0.h,
                                                    (short)0, acc0, false, false);
      acc1 = __builtin_amdgcn_wmma_f32_16x16x32_f16(false, afrag[1][q], false, b1.h,
                                                    (short)0, acc1, false, false);
    }
    const int col = t * 16 + row;    // C/D layout: N = lane&15, M = j + 8*(lane>>4)
#pragma unroll
    for (int j = 0; j < 8; ++j) {
      dst[(j + 8 * kh) * 64 + col] = __logf(acc0[j] * acc1[j]) + ms[j];
    }
  }
}

// ---------------------------------------------------------------------------
// Phase 1: one wave = (16 batch rows) x (16-leaf subtree), tree levels 0..3.
// Depth-first evaluation with a 5-entry LDS stack; result -> INTER.
// ---------------------------------------------------------------------------
__global__ void __launch_bounds__(32) phase1(
    const float* __restrict__ x, const float* __restrict__ mu,
    const float* __restrict__ ga, const float* __restrict__ gb,
    const _Float16* __restrict__ wfh, float* __restrict__ inter)
{
  __shared__ float stk[5][16 * 64];
  __shared__ float mbuf[32];
  __shared__ float msum[16];
  const int lane = threadIdx.x;
  const int bt = blockIdx.x >> 4;     // batch tile
  const int st = blockIdx.x & 15;     // subtree
  const int row = lane & 15, half = lane >> 4;
  const uint4* wf4 = (const uint4*)wfh;

  int sp = 0;
  int lvl[5];
  for (int leaf = 0; leaf < LEAVES_PER_SUB; ++leaf) {
    const int d = st * LEAVES_PER_SUB + leaf;
    // Gaussian leaf: ll[row][k] = -0.5*((x-mu)*exp(-ls))^2 - ls - 0.5*log(2pi)
    const float xv = x[(size_t)(bt * 16 + row) * DDIM + d];
    const float4* m4 = (const float4*)(mu + d * 64 + half * 32);
    const float4* a4 = (const float4*)(ga + d * 64 + half * 32);
    const float4* b4 = (const float4*)(gb + d * 64 + half * 32);
    float4* o4 = (float4*)(stk[sp] + row * 64 + half * 32);
#pragma unroll
    for (int i = 0; i < 8; ++i) {
      float4 m = m4[i], a = a4[i], b = b4[i], r;
      float t;
      t = (xv - m.x) * a.x; r.x = -0.5f * t * t + b.x;
      t = (xv - m.y) * a.y; r.y = -0.5f * t * t + b.y;
      t = (xv - m.z) * a.z; r.z = -0.5f * t * t + b.z;
      t = (xv - m.w) * a.w; r.w = -0.5f * t * t + b.w;
      o4[i] = r;
    }
    lvl[sp++] = 0;
    while (sp >= 2 && lvl[sp - 1] == lvl[sp - 2]) {
      const int l = lvl[sp - 2];                            // 0..3
      const int gbase = st * LEAVES_PER_SUB + ((leaf >> (l + 1)) << (l + 1));
      const int u = lvl_off(l) + (gbase >> (l + 1));
      combine32(stk[sp - 2], stk[sp - 1], stk[sp - 2], mbuf, msum, wf4, u, lane);
      --sp;
      lvl[sp - 1] = l + 1;
    }
  }
  __syncthreads();
  const float4* s4 = (const float4*)(stk[0] + row * 64 + half * 32);
  float4* d4 = (float4*)(inter + ((size_t)(bt * 16 + row) * NSUB + st) * 64 + half * 32);
#pragma unroll
  for (int i = 0; i < 8; ++i) d4[i] = s4[i];
}

// ---------------------------------------------------------------------------
// Phase 2: one wave per batch tile; 16 level-4 regions -> levels 4..7 -> root.
// ---------------------------------------------------------------------------
__global__ void __launch_bounds__(32) phase2(
    const float* __restrict__ inter, const _Float16* __restrict__ wfh,
    const float* __restrict__ lw, float* __restrict__ out)
{
  __shared__ float stk[5][16 * 64];
  __shared__ float mbuf[32];
  __shared__ float msum[16];
  const int lane = threadIdx.x;
  const int bt = blockIdx.x;
  const int row = lane & 15, half = lane >> 4;
  const uint4* wf4 = (const uint4*)wfh;

  int sp = 0;
  int lvl[5];
  for (int leaf = 0; leaf < NSUB; ++leaf) {
    const float4* i4 = (const float4*)(inter + ((size_t)(bt * 16 + row) * NSUB + leaf) * 64 + half * 32);
    float4* o4 = (float4*)(stk[sp] + row * 64 + half * 32);
#pragma unroll
    for (int i = 0; i < 8; ++i) o4[i] = i4[i];
    lvl[sp++] = 0;
    while (sp >= 2 && lvl[sp - 1] == lvl[sp - 2]) {
      const int l = lvl[sp - 2];                 // local 0..3 -> global 4..7
      const int gl = 4 + l;
      const int base = (leaf >> (l + 1)) << (l + 1);   // in level-4-region units
      const int u = lvl_off(gl) + (base >> (l + 1));
      combine32(stk[sp - 2], stk[sp - 1], stk[sp - 2], mbuf, msum, wf4, u, lane);
      --sp;
      lvl[sp - 1] = l + 1;
    }
  }
  __syncthreads();
  // root: out[b] = logsumexp_k( h[b,k] + log_softmax(w_root)[k] )
  if (lane < 16) {
    const float* h = stk[0] + row * 64;
    float mx = -INFINITY;
    for (int k = 0; k < 64; ++k) mx = fmaxf(mx, h[k] + lw[k]);
    float s = 0.f;
    for (int k = 0; k < 64; ++k) s += __expf(h[k] + lw[k] - mx);
    out[bt * 16 + row] = mx + __logf(s);
  }
}

// ---------------------------------------------------------------------------
extern "C" void kernel_launch(void* const* d_in, const int* in_sizes, int n_in,
                              void* d_out, int out_size, void* d_ws, size_t ws_size,
                              hipStream_t stream) {
  (void)in_sizes; (void)n_in; (void)out_size; (void)ws_size;
  const float* x  = (const float*)d_in[0];
  const float* mu = (const float*)d_in[1];
  const float* ls = (const float*)d_in[2];
  W8 w;
  for (int i = 0; i < 8; ++i) w.p[i] = (const float*)d_in[3 + i];
  const float* wroot = (const float*)d_in[11];

  char* ws = (char*)d_ws;
  _Float16* wf = (_Float16*)(ws + WFRAG_OFF);
  float* ga    = (float*)(ws + GA_OFF);
  float* gb    = (float*)(ws + GB_OFF);
  float* lw    = (float*)(ws + LW_OFF);
  float* inter = (float*)(ws + INTER_OFF);
  float* out   = (float*)d_out;

  prep_weights<<<dim3(510), dim3(64), 0, stream>>>(w, wf);
  prep_gauss<<<dim3(64), dim3(256), 0, stream>>>(ls, ga, gb);
  prep_root<<<dim3(1), dim3(64), 0, stream>>>(wroot, lw);
  phase1<<<dim3(NTILES_B * NSUB), dim3(32), 0, stream>>>(x, mu, ga, gb, wf, inter);
  phase2<<<dim3(NTILES_B), dim3(32), 0, stream>>>(inter, wf, lw, out);
}
